// MultiQueryAttention_59562606460992
// MI455X (gfx1250) — compile-verified, matrix operand order checked by reading, not run on hardware
//
#include <hip/hip_runtime.h>
#include <cstdint>
#include <cstddef>

// ---------------- problem constants (fixed by the reference) ----------------
#define B_   2
#define S_   2048
#define H_   2048
#define NH_  16
#define NG_  4
#define HPG_ 4
#define HD_  128
#define KV_  (NG_ * HD_)   // 512
#define M_   (B_ * S_)     // 4096 rows for the projections

// ---------------- WMMA fragment helpers (bf16 16x16x32, f32 accum) ----------
typedef __attribute__((ext_vector_type(16))) __bf16 v16bf;
typedef __attribute__((ext_vector_type(8)))  float  v8f;

union ABFrag {
  v16bf v;
  unsigned short s[16];
  uint4 q[2];
};

static __device__ __forceinline__ unsigned short f32_to_bf16(float f) {
  union { float f; unsigned int u; } c; c.f = f;
  unsigned int u = c.u;
  u += 0x7FFFu + ((u >> 16) & 1u);          // round-to-nearest-even
  return (unsigned short)(u >> 16);
}

// A/B fragment loader for 16-bit 16x32 operands (CDNA5 ISA 7.12.2):
// lane l (l<16): row = l,   K = {kofs+0..7,  kofs+16..23}
// lane l (>=16): row = l-16,K = {kofs+8..15, kofs+24..31}
static __device__ __forceinline__ void load_frag(ABFrag& f, const unsigned short* base,
                                                 int ld, int row16, int kofs, int lane) {
  const unsigned short* p =
      base + (size_t)(row16 + (lane & 15)) * ld + kofs + ((lane >> 4) << 3);
  f.q[0] = *(const uint4*)(p);
  f.q[1] = *(const uint4*)(p + 16);
}

static __device__ __forceinline__ v8f wmma_bf16(const ABFrag& a, const ABFrag& b, v8f c) {
  return __builtin_amdgcn_wmma_f32_16x16x32_bf16(false, a.v, false, b.v,
                                                 (short)0, c, false, false);
}

// ---------------- Tensor Data Mover: 2D tile load, global -> LDS ------------
// Builds a D# per CDNA5 ISA ch.8 and issues TENSOR_LOAD_TO_LDS.
// pad_interval code: 0=2,1=4,2=8,3=16,4=32,5=64 DWORDs; pad_amount code: n -> n+1 DWORDs.
#define HAS_TDM __has_builtin(__builtin_amdgcn_tensor_load_to_lds)

#if HAS_TDM
typedef unsigned __attribute__((ext_vector_type(4))) v4u_;
typedef int      __attribute__((ext_vector_type(8))) v8i_;
typedef int      __attribute__((ext_vector_type(4))) v4i_;

static __device__ __forceinline__ void tdm_load_2d(
    unsigned lds_off, unsigned long long gaddr,
    unsigned tensor_d0, unsigned tensor_d1,
    unsigned tile_d0, unsigned tile_d1, unsigned stride0_elems,
    unsigned pad_interval_code, unsigned pad_amount_code) {
  v4u_ g0;
  g0[0] = 1u;                                   // count=1, user descriptor
  g0[1] = lds_off;                              // lds_addr (bytes)
  g0[2] = (unsigned)gaddr;                      // global_addr[31:0]
  g0[3] = (unsigned)(gaddr >> 32) | (2u << 30); // global_addr[56:32] | type=2
  v8i_ g1;
  g1[0] = (int)((1u << 16)                      // data_size = 1 (2 bytes, bf16)
              | (1u << 20)                      // pad_enable
              | (pad_interval_code << 22)
              | (pad_amount_code << 25));
  g1[1] = (int)((tensor_d0 & 0xFFFFu) << 16);                       // dim0[15:0]
  g1[2] = (int)((tensor_d0 >> 16) | ((tensor_d1 & 0xFFFFu) << 16)); // dim0 hi, dim1 lo
  g1[3] = (int)((tensor_d1 >> 16) | (tile_d0 << 16));               // dim1 hi, tile0
  g1[4] = (int)(tile_d1);                                           // tile1, tile2=0
  g1[5] = (int)stride0_elems;                                       // dim0_stride lo
  g1[6] = 0;
  g1[7] = 0;
  v4i_ z4 = {0, 0, 0, 0};
  v8i_ z8 = {0, 0, 0, 0, 0, 0, 0, 0};
  // clang-23 / therock-10.0 form: (g0, g1, g2, g3, g4, cpol)
  __builtin_amdgcn_tensor_load_to_lds(g0, g1, z4, z4, z8, 0);
}
#endif

// ---------------- kernel 1: bf16 Out = (X @ W^T + bias) * outScale ----------
// MODE 0: row-major bf16 Out[M,N].
// MODE 1: per-head transposed bf16 (for V):  Out[((b*NG+g)*HD+d)*S + t]
#define PJ_BM 128
#define PJ_BN 64
#define PJ_BK 32

template <int MODE>
__global__ __launch_bounds__(256)
void qkv_proj_bf16(const float* __restrict__ X, const float* __restrict__ W,
                   const float* __restrict__ bias, unsigned short* __restrict__ Out,
                   int M, int N, int K, float outScale) {
  __shared__ __align__(16) unsigned short sX[PJ_BM][PJ_BK + 8];
  __shared__ __align__(16) unsigned short sW[PJ_BN][PJ_BK + 8];

  const int tid  = threadIdx.x;
  const int lane = tid & 31;
  const int wv   = tid >> 5;
  const int wm   = wv & 3;
  const int wn   = wv >> 2;
  const int m0   = blockIdx.y * PJ_BM;
  const int n0   = blockIdx.x * PJ_BN;

  v8f C[2][2];
#pragma unroll
  for (int i = 0; i < 2; ++i)
#pragma unroll
    for (int j = 0; j < 2; ++j)
      C[i][j] = (v8f){0.f, 0.f, 0.f, 0.f, 0.f, 0.f, 0.f, 0.f};

  for (int k0 = 0; k0 < K; k0 += PJ_BK) {
#pragma unroll
    for (int i = 0; i < 4; ++i) {                  // X tile 128x32
      int idx = tid + i * 256;
      int row = idx >> 3;
      int c4  = (idx & 7) * 4;
      float4 f = *(const float4*)(X + (size_t)(m0 + row) * K + k0 + c4);
      sX[row][c4 + 0] = f32_to_bf16(f.x);
      sX[row][c4 + 1] = f32_to_bf16(f.y);
      sX[row][c4 + 2] = f32_to_bf16(f.z);
      sX[row][c4 + 3] = f32_to_bf16(f.w);
    }
#pragma unroll
    for (int i = 0; i < 2; ++i) {                  // W tile 64x32
      int idx = tid + i * 256;
      int row = idx >> 3;
      int c4  = (idx & 7) * 4;
      float4 f = *(const float4*)(W + (size_t)(n0 + row) * K + k0 + c4);
      sW[row][c4 + 0] = f32_to_bf16(f.x);
      sW[row][c4 + 1] = f32_to_bf16(f.y);
      sW[row][c4 + 2] = f32_to_bf16(f.z);
      sW[row][c4 + 3] = f32_to_bf16(f.w);
    }
    __syncthreads();

    ABFrag A[2], Bf[2];
#pragma unroll
    for (int i = 0; i < 2; ++i)
      load_frag(A[i], &sX[0][0], PJ_BK + 8, wm * 32 + i * 16, 0, lane);
#pragma unroll
    for (int j = 0; j < 2; ++j)
      load_frag(Bf[j], &sW[0][0], PJ_BK + 8, wn * 32 + j * 16, 0, lane);
#pragma unroll
    for (int i = 0; i < 2; ++i)
#pragma unroll
      for (int j = 0; j < 2; ++j)
        C[i][j] = wmma_bf16(A[i], Bf[j], C[i][j]);
    __syncthreads();
  }

  const int mBase = m0 + wm * 32;
  const int nBase = n0 + wn * 32;
#pragma unroll
  for (int i = 0; i < 2; ++i)
#pragma unroll
    for (int j = 0; j < 2; ++j)
#pragma unroll
      for (int r = 0; r < 8; ++r) {
        int row = mBase + i * 16 + ((lane & 16) ? (8 + r) : r);
        int col = nBase + j * 16 + (lane & 15);
        float v = (C[i][j][r] + bias[col]) * outScale;
        if (MODE == 0) {
          Out[(size_t)row * N + col] = f32_to_bf16(v);
        } else {
          int bb = row >> 11, t = row & (S_ - 1);     // S_ = 2048
          int gg = col >> 7,  d = col & (HD_ - 1);
          Out[((size_t)((bb * NG_ + gg) * HD_ + d)) * S_ + t] = f32_to_bf16(v);
        }
      }
}

// ---------------- kernel 2: fused flash GQA attention -----------------------
// One block per (b, head, 64 query rows). 128 threads = 4 waves, each wave
// owns 16 q-rows x HD=128. K and V^T tiles arrive via the Tensor Data Mover.
__global__ __launch_bounds__(128)
void mqa_attn(const unsigned short* __restrict__ Q,   // [B*S, H] bf16, pre-scaled
              const unsigned short* __restrict__ Kb,  // [B*S, KV_] bf16
              const unsigned short* __restrict__ VT,  // [B*NG*HD, S] bf16 (transposed)
              const int*            __restrict__ amask, // [B, S]
              float* __restrict__ Out) {              // [B*S, H] f32
  __shared__ __align__(16) unsigned short sK [64][HD_ + 8];   // [key][d]
  __shared__ __align__(16) unsigned short sVT[HD_][64 + 8];   // [d][key]
  __shared__ __align__(16) unsigned short sP [4][16][64 + 8]; // per-wave P
  __shared__ float sMask[64];

  const int tid  = threadIdx.x;
  const int lane = tid & 31;
  const int wv   = tid >> 5;
  const int b    = blockIdx.z;
  const int h    = blockIdx.y;
  const int g    = h >> 2;
  const int qRow0 = blockIdx.x * 64 + wv * 16;

  // ---- Q fragments straight from global bf16 (scale folded at projection)
  ABFrag QA[4];
  {
    int row = qRow0 + (lane & 15);
    const unsigned short* qrow = Q + ((size_t)(b * S_) + row) * H_ + h * HD_;
    int kh = (lane >> 4) * 8;
#pragma unroll
    for (int s = 0; s < 4; ++s) {
      int d0 = s * 32 + kh;
      QA[s].q[0] = *(const uint4*)(qrow + d0);
      QA[s].q[1] = *(const uint4*)(qrow + d0 + 16);
    }
  }

  float mrow[8], lrow[8];
  v8f   Oacc[8];
#pragma unroll
  for (int r = 0; r < 8; ++r) { mrow[r] = -1e30f; lrow[r] = 0.f; }
#pragma unroll
  for (int dt = 0; dt < 8; ++dt)
    Oacc[dt] = (v8f){0.f, 0.f, 0.f, 0.f, 0.f, 0.f, 0.f, 0.f};

  const int NT = S_ / 64;
  for (int nt = 0; nt < NT; ++nt) {
    const int kt0 = nt * 64;

#if HAS_TDM
    if (wv == 0) {
      // K tile: 64 rows x 128 bf16; LDS rows padded 128->136 halves
      // row = 256B = 64 DW -> pad_interval code 5; pad 16B = 4 DW -> code 3
      tdm_load_2d((unsigned)(size_t)&sK[0][0],
                  (unsigned long long)(size_t)(Kb + ((size_t)(b * S_) + kt0) * KV_ + g * HD_),
                  /*tensor_d0=*/KV_, /*tensor_d1=*/M_,
                  /*tile_d0=*/HD_, /*tile_d1=*/64, /*stride0=*/KV_,
                  /*pad_interval=*/5u, /*pad_amount=*/3u);
      // V^T tile: 128 rows x 64 bf16; LDS rows padded 64->72 halves
      // row = 128B = 32 DW -> code 4; pad 4 DW -> code 3
      tdm_load_2d((unsigned)(size_t)&sVT[0][0],
                  (unsigned long long)(size_t)(VT + ((size_t)((b * NG_ + g) * HD_)) * S_ + kt0),
                  /*tensor_d0=*/S_, /*tensor_d1=*/(unsigned)(B_ * NG_ * HD_),
                  /*tile_d0=*/64, /*tile_d1=*/HD_, /*stride0=*/S_,
                  /*pad_interval=*/4u, /*pad_amount=*/3u);
      __builtin_amdgcn_s_wait_tensorcnt(0);
    }
#else
    // Fallback: cooperative bf16 copies (8x uint4 per thread per tile)
#pragma unroll
    for (int i = 0; i < 8; ++i) {
      int idx = tid + i * 128;            // 0..1023 uint4 slots
      int key = idx >> 4;
      int c8  = (idx & 15) * 8;
      *(uint4*)&sK[key][c8] =
          *(const uint4*)(Kb + ((size_t)(b * S_) + kt0 + key) * KV_ + g * HD_ + c8);
    }
#pragma unroll
    for (int i = 0; i < 8; ++i) {
      int idx = tid + i * 128;
      int d   = idx >> 3;
      int c8  = (idx & 7) * 8;
      *(uint4*)&sVT[d][c8] =
          *(const uint4*)(VT + ((size_t)((b * NG_ + g) * HD_ + d)) * S_ + kt0 + c8);
    }
#endif
    if (tid < 64) {
      int t = kt0 + tid;
      sMask[tid] = amask[b * S_ + t] ? 0.0f : -1e30f;
    }
    __syncthreads();

    // ---- scores: 4 tiles of 16x16 over HD (4 K-steps of 32)
    v8f sc[4];
#pragma unroll
    for (int jn = 0; jn < 4; ++jn) {
      v8f c = (v8f){0.f, 0.f, 0.f, 0.f, 0.f, 0.f, 0.f, 0.f};
#pragma unroll
      for (int s = 0; s < 4; ++s) {
        ABFrag Bf;
        load_frag(Bf, &sK[0][0], HD_ + 8, jn * 16, s * 32, lane);
        c = wmma_bf16(QA[s], Bf, c);
      }
      sc[jn] = c;
    }

    // ---- masked online softmax per fragment row
    const int colL = lane & 15;
#pragma unroll
    for (int r = 0; r < 8; ++r) {
      float x0 = sc[0][r] + sMask[colL];
      float x1 = sc[1][r] + sMask[16 + colL];
      float x2 = sc[2][r] + sMask[32 + colL];
      float x3 = sc[3][r] + sMask[48 + colL];
      float xm = fmaxf(fmaxf(x0, x1), fmaxf(x2, x3));
#pragma unroll
      for (int off = 1; off < 16; off <<= 1)
        xm = fmaxf(xm, __shfl_xor(xm, off, 32));
      float mnew  = fmaxf(mrow[r], xm);
      float alpha = __expf(mrow[r] - mnew);
      mrow[r] = mnew;
      float p0 = __expf(x0 - mnew), p1 = __expf(x1 - mnew);
      float p2 = __expf(x2 - mnew), p3 = __expf(x3 - mnew);
      float rs = p0 + p1 + p2 + p3;
#pragma unroll
      for (int off = 1; off < 16; off <<= 1)
        rs += __shfl_xor(rs, off, 32);
      lrow[r] = lrow[r] * alpha + rs;
      int prow = (lane & 16) ? (8 + r) : r;
      sP[wv][prow][colL]      = f32_to_bf16(p0);
      sP[wv][prow][16 + colL] = f32_to_bf16(p1);
      sP[wv][prow][32 + colL] = f32_to_bf16(p2);
      sP[wv][prow][48 + colL] = f32_to_bf16(p3);
#pragma unroll
      for (int dt = 0; dt < 8; ++dt) Oacc[dt][r] *= alpha;
    }
    __syncthreads();

    // ---- O += P @ V
    ABFrag Pa0, Pa1;
    load_frag(Pa0, &sP[wv][0][0], 64 + 8, 0, 0, lane);
    load_frag(Pa1, &sP[wv][0][0], 64 + 8, 0, 32, lane);
#pragma unroll
    for (int dt = 0; dt < 8; ++dt) {
      ABFrag Vb0, Vb1;
      load_frag(Vb0, &sVT[0][0], 64 + 8, dt * 16, 0, lane);
      Oacc[dt] = wmma_bf16(Pa0, Vb0, Oacc[dt]);
      load_frag(Vb1, &sVT[0][0], 64 + 8, dt * 16, 32, lane);
      Oacc[dt] = wmma_bf16(Pa1, Vb1, Oacc[dt]);
    }
    __syncthreads();
  }

  // ---- normalize and write f32 output
#pragma unroll
  for (int r = 0; r < 8; ++r) {
    float inv = 1.0f / lrow[r];
    int orow  = qRow0 + ((lane & 16) ? (8 + r) : r);
    float* op = Out + ((size_t)(b * S_) + orow) * H_ + h * HD_;
#pragma unroll
    for (int dt = 0; dt < 8; ++dt) {
      int col = dt * 16 + (lane & 15);
      op[col] = Oacc[dt][r] * inv;
    }
  }
}

// ---------------- host launcher ---------------------------------------------
extern "C" void kernel_launch(void* const* d_in, const int* in_sizes, int n_in,
                              void* d_out, int out_size, void* d_ws, size_t ws_size,
                              hipStream_t stream) {
  const float* hs    = (const float*)d_in[0];
  const int*   amask = (const int*)  d_in[1];
  const float* Wq    = (const float*)d_in[2];
  const float* bq    = (const float*)d_in[3];
  const float* Wk    = (const float*)d_in[4];
  const float* bk    = (const float*)d_in[5];
  const float* Wv    = (const float*)d_in[6];
  const float* bv    = (const float*)d_in[7];
  float* out = (float*)d_out;

  const size_t qElems = (size_t)M_ * H_;    // bf16
  const size_t kElems = (size_t)M_ * KV_;   // bf16
  const size_t need   = (qElems + 2 * kElems) * sizeof(unsigned short); // ~25 MB
  if (ws_size < need) return;

  unsigned short* qbf = (unsigned short*)d_ws;
  unsigned short* kbf = qbf + qElems;
  unsigned short* vT  = kbf + kElems;

  const float scale = 0.08838834764831845f; // 1/sqrt(128), folded into Q

  qkv_proj_bf16<0><<<dim3(H_  / PJ_BN, M_ / PJ_BM), dim3(256), 0, stream>>>(
      hs, Wq, bq, qbf, M_, H_,  H_, scale);
  qkv_proj_bf16<0><<<dim3(KV_ / PJ_BN, M_ / PJ_BM), dim3(256), 0, stream>>>(
      hs, Wk, bk, kbf, M_, KV_, H_, 1.0f);
  qkv_proj_bf16<1><<<dim3(KV_ / PJ_BN, M_ / PJ_BM), dim3(256), 0, stream>>>(
      hs, Wv, bv, vT,  M_, KV_, H_, 1.0f);

  mqa_attn<<<dim3(S_ / 64, NH_, B_), dim3(128), 0, stream>>>(
      qbf, kbf, vT, amask, out);
}
